// SimpleTransformer_69561290326689
// MI455X (gfx1250) — matrix operand order repaired and checked by reading, not compiled, not run on hardware
//
#include <hip/hip_runtime.h>
#include <hip/hip_bf16.h>
#include <math.h>

typedef __bf16 bf16_t;
typedef __attribute__((ext_vector_type(16))) __bf16 v16bf;
typedef __attribute__((ext_vector_type(8)))  float   v8f;
typedef __attribute__((ext_vector_type(4)))  unsigned u32x4_t;
typedef __attribute__((ext_vector_type(8)))  int      i32x8_t;
typedef __attribute__((ext_vector_type(4)))  int      i32x4_t;

#define DEVI static __device__ __forceinline__

constexpr int Bc = 4, Tc = 512, Cc = 512, Ic = 2048, Ec = 32, Vc = 32000, Hc = 8;
constexpr int Dc = Cc / Hc;   // 64
constexpr int Nc = Bc * Tc;   // 2048

// ---------------------------------------------------------------------------
// WMMA helpers (CDNA5 wave32, 16x16x32 bf16 -> f32)
// ---------------------------------------------------------------------------
DEVI v8f wmma_bf16(v16bf a, v16bf b, v8f c) {
  return __builtin_amdgcn_wmma_f32_16x16x32_bf16(false, a, false, b, (short)0, c,
                                                 false, false);
}

// A fragment 16(M) x 32(K), row-major source, bf16 (contiguous k = 16*half + i, matching B).
DEVI v16bf a_frag_bf16(const bf16_t* base, size_t ld, int k0) {
  const int lane = threadIdx.x & 31;
  const int row  = lane & 15;
  const int kh   = (lane < 16) ? 0 : 16;
  v16bf a;
  const bf16_t* p = base + (size_t)row * ld + k0 + kh;
#pragma unroll
  for (int i = 0; i < 8; ++i) a[i] = p[i];
#pragma unroll
  for (int i = 0; i < 8; ++i) a[8 + i] = p[8 + i];
  return a;
}

// B fragment 32(K) x 16(N), row-major K x N source, f32 converted to bf16.
DEVI v16bf b_frag_f32(const float* base, size_t ld, int k0, int n0) {
  const int lane = threadIdx.x & 31;
  const int col  = lane & 15;
  const int kr   = k0 + ((lane < 16) ? 0 : 16);
  v16bf b;
  const float* p = base + (size_t)kr * ld + n0 + col;
#pragma unroll
  for (int i = 0; i < 16; ++i) b[i] = (bf16_t)p[(size_t)i * ld];
  return b;
}

DEVI v16bf b_frag_bf16(const bf16_t* base, size_t ld, int k0, int n0) {
  const int lane = threadIdx.x & 31;
  const int col  = lane & 15;
  const int kr   = k0 + ((lane < 16) ? 0 : 16);
  v16bf b;
  const bf16_t* p = base + (size_t)kr * ld + n0 + col;
#pragma unroll
  for (int i = 0; i < 16; ++i) b[i] = p[(size_t)i * ld];
  return b;
}

// B fragment where memory holds N x K (we need B = M^T). Used for Q*K^T.
DEVI v16bf bT_frag_bf16(const bf16_t* base, size_t ld, int k0, int n0) {
  const int lane = threadIdx.x & 31;
  const int n    = n0 + (lane & 15);
  const int kr   = k0 + ((lane < 16) ? 0 : 16);
  v16bf b;
  const bf16_t* p = base + (size_t)n * ld + kr;
#pragma unroll
  for (int i = 0; i < 16; ++i) b[i] = p[i];
  return b;
}

DEVI void store_tile_f32(float* base, size_t ld, const v8f& d) {
  const int lane = threadIdx.x & 31;
  const int col  = lane & 15;
  const int r0   = (lane < 16) ? 0 : 8;
#pragma unroll
  for (int i = 0; i < 8; ++i) base[(size_t)(r0 + i) * ld + col] = d[i];
}

DEVI void store_tile_bf16(bf16_t* base, size_t ld, const v8f& d) {
  const int lane = threadIdx.x & 31;
  const int col  = lane & 15;
  const int r0   = (lane < 16) ? 0 : 8;
#pragma unroll
  for (int i = 0; i < 8; ++i) base[(size_t)(r0 + i) * ld + col] = (bf16_t)d[i];
}

DEVI float half16_max(float v) {
#pragma unroll
  for (int o = 1; o < 16; o <<= 1) v = fmaxf(v, __shfl_xor(v, o, 32));
  return v;
}
DEVI float half16_sum(float v) {
#pragma unroll
  for (int o = 1; o < 16; o <<= 1) v += __shfl_xor(v, o, 32);
  return v;
}

// ---------------------------------------------------------------------------
// CDNA5 async global->LDS copy (ASYNCcnt path, ISA 15.18.3 op 98)
// ---------------------------------------------------------------------------
DEVI void async_copy_b128(void* lds_dst, const void* gsrc) {
  unsigned loff = (unsigned)(uintptr_t)lds_dst;            // LDS byte offset
  unsigned long long g = (unsigned long long)(uintptr_t)gsrc;
  asm volatile("global_load_async_to_lds_b128 %0, %1, off"
               :: "v"(loff), "v"(g) : "memory");
}
DEVI void async_wait0() {
  asm volatile("s_wait_asynccnt 0x0" ::: "memory");
}

// stage a 16 x Cc bf16 A tile (16 KB) into LDS, shared by all waves of block
DEVI void stage_a_async(bf16_t* dst, const bf16_t* src) {
  for (int idx = threadIdx.x; idx < (16 * Cc) / 8; idx += blockDim.x)
    async_copy_b128(dst + idx * 8, src + idx * 8);
  async_wait0();
}

// ---------------------------------------------------------------------------
// Utility kernels
// ---------------------------------------------------------------------------
__global__ void k_zero_f32(float* p, size_t n) {
  size_t i = (size_t)blockIdx.x * blockDim.x + threadIdx.x;
  size_t stride = (size_t)gridDim.x * blockDim.x;
  for (; i < n; i += stride) p[i] = 0.0f;
}

__global__ void k_zero_i32(int* p, int n) {
  int i = blockIdx.x * blockDim.x + threadIdx.x;
  if (i < n) p[i] = 0;
}

// embedding gather + LayerNorm1 -> h (f32) and x1 (bf16)
__global__ void k_embed_ln1(const int* __restrict__ ids,
                            const float* __restrict__ emb,
                            const float* __restrict__ g,
                            const float* __restrict__ b,
                            float* __restrict__ h, bf16_t* __restrict__ xb) {
  const int t   = blockIdx.x;
  const int tid = threadIdx.x;     // 256 threads, 2 elems each
  const float* row = emb + (size_t)ids[t] * Cc;
  float v0 = row[tid], v1 = row[tid + 256];
  h[(size_t)t * Cc + tid]       = v0;
  h[(size_t)t * Cc + tid + 256] = v1;
  __shared__ float ssum[256], ssq[256];
  ssum[tid] = v0 + v1;
  ssq[tid]  = v0 * v0 + v1 * v1;
  __syncthreads();
  for (int s = 128; s > 0; s >>= 1) {
    if (tid < s) { ssum[tid] += ssum[tid + s]; ssq[tid] += ssq[tid + s]; }
    __syncthreads();
  }
  const float mu  = ssum[0] * (1.0f / Cc);
  const float var = ssq[0] * (1.0f / Cc) - mu * mu;
  const float inv = rsqrtf(var + 1e-5f);
  xb[(size_t)t * Cc + tid]       = (bf16_t)((v0 - mu) * inv * g[tid] + b[tid]);
  xb[(size_t)t * Cc + tid + 256] = (bf16_t)((v1 - mu) * inv * g[tid + 256] + b[tid + 256]);
}

// LayerNorm2: h -> x2 (f32, model output) and x2b (bf16)
__global__ void k_ln2(const float* __restrict__ h, const float* __restrict__ g,
                      const float* __restrict__ b, float* __restrict__ x2,
                      bf16_t* __restrict__ x2b) {
  const int t   = blockIdx.x;
  const int tid = threadIdx.x;
  float v0 = h[(size_t)t * Cc + tid], v1 = h[(size_t)t * Cc + tid + 256];
  __shared__ float ssum[256], ssq[256];
  ssum[tid] = v0 + v1;
  ssq[tid]  = v0 * v0 + v1 * v1;
  __syncthreads();
  for (int s = 128; s > 0; s >>= 1) {
    if (tid < s) { ssum[tid] += ssum[tid + s]; ssq[tid] += ssq[tid + s]; }
    __syncthreads();
  }
  const float mu  = ssum[0] * (1.0f / Cc);
  const float var = ssq[0] * (1.0f / Cc) - mu * mu;
  const float inv = rsqrtf(var + 1e-5f);
  float o0 = (v0 - mu) * inv * g[tid] + b[tid];
  float o1 = (v1 - mu) * inv * g[tid + 256] + b[tid + 256];
  x2[(size_t)t * Cc + tid]        = o0;
  x2[(size_t)t * Cc + tid + 256]  = o1;
  x2b[(size_t)t * Cc + tid]       = (bf16_t)o0;
  x2b[(size_t)t * Cc + tid + 256] = (bf16_t)o1;
}

// ---------------------------------------------------------------------------
// QKV projection: q/k/v = x1 @ W ; A tile staged in LDS via async copy
// grid = (Cc/16/8, Nc/16, 3), block = 256 (8 waves)
// ---------------------------------------------------------------------------
__global__ void k_qkv(const bf16_t* __restrict__ xb, const float* __restrict__ wq,
                      const float* __restrict__ wk, const float* __restrict__ wv,
                      bf16_t* __restrict__ qb, bf16_t* __restrict__ kb,
                      bf16_t* __restrict__ vb) {
  __shared__ bf16_t at[16 * Cc];            // 16 KB A tile
  const int m0 = blockIdx.y * 16;
  stage_a_async(at, xb + (size_t)m0 * Cc);
  __syncthreads();

  const int wave = threadIdx.x >> 5;
  const int n0 = (blockIdx.x * 8 + wave) * 16;
  const float* W = (blockIdx.z == 0) ? wq : (blockIdx.z == 1) ? wk : wv;
  bf16_t* O = (blockIdx.z == 0) ? qb : (blockIdx.z == 1) ? kb : vb;
  v8f acc = {};
#pragma unroll 4
  for (int k0 = 0; k0 < Cc; k0 += 32) {
    v16bf a = a_frag_bf16(at, Cc, k0);
    v16bf b = b_frag_f32(W, Cc, k0, n0);
    acc = wmma_bf16(a, b, acc);
  }
  store_tile_bf16(O + (size_t)m0 * Cc + n0, Cc, acc);
}

// ---------------------------------------------------------------------------
// Flash attention, one wave per (b,h, 16-query tile), causal, WMMA scores+PV
// grid = (Tc/16, Bc*Hc), block = 32
// ---------------------------------------------------------------------------
__global__ void k_attn(const bf16_t* __restrict__ qb, const bf16_t* __restrict__ kb,
                       const bf16_t* __restrict__ vb, bf16_t* __restrict__ aob) {
  const int bh  = blockIdx.y;
  const int bb  = bh / Hc, hh = bh % Hc;
  const int m0  = blockIdx.x * 16;
  const int hd0 = hh * Dc;
  const int lane = threadIdx.x & 31;
  const int col  = lane & 15;
  const int r0   = (lane < 16) ? 0 : 8;

  const bf16_t* Q = qb + ((size_t)(bb * Tc + m0)) * Cc + hd0;
  v16bf qa0 = a_frag_bf16(Q, Cc, 0);
  v16bf qa1 = a_frag_bf16(Q, Cc, 32);

  v8f o0 = {}, o1 = {}, o2 = {}, o3 = {};
  float rmax[8], rsum[8];
#pragma unroll
  for (int i = 0; i < 8; ++i) { rmax[i] = -3.0e38f; rsum[i] = 0.0f; }

  __shared__ bf16_t pl[16 * 32];   // probabilities bounce buffer (A-layout)

  for (int j0 = 0; j0 <= m0 + 15; j0 += 32) {
    const bf16_t* Kb = kb + ((size_t)(bb * Tc + j0)) * Cc + hd0;
    v8f s_lo = {}, s_hi = {};
    s_lo = wmma_bf16(qa0, bT_frag_bf16(Kb, Cc, 0, 0),  s_lo);
    s_lo = wmma_bf16(qa1, bT_frag_bf16(Kb, Cc, 32, 0), s_lo);
    s_hi = wmma_bf16(qa0, bT_frag_bf16(Kb, Cc, 0, 16), s_hi);
    s_hi = wmma_bf16(qa1, bT_frag_bf16(Kb, Cc, 32, 16), s_hi);

#pragma unroll
    for (int i = 0; i < 8; ++i) {
      const int qrow = m0 + r0 + i;
      float v0 = s_lo[i] * 0.125f;               // 1/sqrt(64)
      float v1 = s_hi[i] * 0.125f;
      if (j0 + col > qrow)      v0 = -3.0e38f;   // causal mask
      if (j0 + 16 + col > qrow) v1 = -3.0e38f;
      float bm = half16_max(fmaxf(v0, v1));
      float newm = fmaxf(rmax[i], bm);
      float alpha = __expf(rmax[i] - newm);
      float p0 = __expf(v0 - newm);
      float p1 = __expf(v1 - newm);
      rsum[i] = rsum[i] * alpha + half16_sum(p0 + p1);
      rmax[i] = newm;
      pl[(r0 + i) * 32 + col]      = (bf16_t)p0;
      pl[(r0 + i) * 32 + 16 + col] = (bf16_t)p1;
      o0[i] *= alpha; o1[i] *= alpha; o2[i] *= alpha; o3[i] *= alpha;
    }
    __syncthreads();
    v16bf pa = a_frag_bf16(pl, 32, 0);
    const bf16_t* Vb = vb + ((size_t)(bb * Tc + j0)) * Cc + hd0;
    o0 = wmma_bf16(pa, b_frag_bf16(Vb, Cc, 0, 0),  o0);
    o1 = wmma_bf16(pa, b_frag_bf16(Vb, Cc, 0, 16), o1);
    o2 = wmma_bf16(pa, b_frag_bf16(Vb, Cc, 0, 32), o2);
    o3 = wmma_bf16(pa, b_frag_bf16(Vb, Cc, 0, 48), o3);
    __syncthreads();
  }

#pragma unroll
  for (int i = 0; i < 8; ++i) {
    const float inv = 1.0f / rsum[i];
    o0[i] *= inv; o1[i] *= inv; o2[i] *= inv; o3[i] *= inv;
  }
  bf16_t* O = aob + ((size_t)(bb * Tc + m0)) * Cc + hd0;
  store_tile_bf16(O + 0,  Cc, o0);
  store_tile_bf16(O + 16, Cc, o1);
  store_tile_bf16(O + 32, Cc, o2);
  store_tile_bf16(O + 48, Cc, o3);
}

// ---------------------------------------------------------------------------
// O projection + residual: h += ao @ wo ; A tile staged in LDS via async copy
// ---------------------------------------------------------------------------
__global__ void k_oproj(const bf16_t* __restrict__ aob, const float* __restrict__ wo,
                        float* __restrict__ h) {
  __shared__ bf16_t at[16 * Cc];
  const int m0 = blockIdx.y * 16;
  stage_a_async(at, aob + (size_t)m0 * Cc);
  __syncthreads();

  const int wave = threadIdx.x >> 5;
  const int n0 = (blockIdx.x * 8 + wave) * 16;
  v8f acc = {};
#pragma unroll 4
  for (int k0 = 0; k0 < Cc; k0 += 32) {
    v16bf a = a_frag_bf16(at, Cc, k0);
    v16bf b = b_frag_f32(wo, Cc, k0, n0);
    acc = wmma_bf16(a, b, acc);
  }
  float* Hb = h + (size_t)m0 * Cc + n0;
  const int lane = threadIdx.x & 31;
  const int col  = lane & 15;
  const int r0   = (lane < 16) ? 0 : 8;
#pragma unroll
  for (int i = 0; i < 8; ++i) Hb[(size_t)(r0 + i) * Cc + col] += acc[i];
}

// ---------------------------------------------------------------------------
// Router logits: x2 @ gate -> router_logits [N,32] (model output)
// grid = Nc/16, block = 64 (2 waves, one per 16-col tile)
// ---------------------------------------------------------------------------
__global__ void k_router(const bf16_t* __restrict__ x2b, const float* __restrict__ gate,
                         float* __restrict__ rl) {
  const int wave = threadIdx.x >> 5;
  const int n0 = wave * 16;
  const int m0 = blockIdx.x * 16;
  v8f acc = {};
  for (int k0 = 0; k0 < Cc; k0 += 32) {
    v16bf a = a_frag_bf16(x2b + (size_t)m0 * Cc, Cc, k0);
    v16bf b = b_frag_f32(gate, Ec, k0, n0);
    acc = wmma_bf16(a, b, acc);
  }
  store_tile_f32(rl + (size_t)m0 * Ec + n0, Ec, acc);
}

// top-2 + softmax weights + per-expert token gather
__global__ void k_topk(const float* __restrict__ rl, float* __restrict__ rw,
                       int* __restrict__ sel, int* __restrict__ counts,
                       int* __restrict__ list) {
  const int t = blockIdx.x * blockDim.x + threadIdx.x;
  if (t >= Nc) return;
  const float* r = rl + (size_t)t * Ec;
  int i0 = 0; float v0 = r[0];
  for (int e = 1; e < Ec; ++e) { float v = r[e]; if (v > v0) { v0 = v; i0 = e; } }
  int i1 = (i0 == 0) ? 1 : 0; float v1 = r[i1];
  for (int e = 0; e < Ec; ++e) {
    if (e == i0) continue;
    float v = r[e]; if (v > v1) { v1 = v; i1 = e; }
  }
  const float ex = __expf(v1 - v0);
  const float inv = 1.0f / (1.0f + ex);
  rw[t * 2] = inv; rw[t * 2 + 1] = ex * inv;
  sel[t * 2] = i0; sel[t * 2 + 1] = i1;
  int s0 = atomicAdd(&counts[i0], 1); list[(size_t)i0 * Nc + s0] = t;
  int s1 = atomicAdd(&counts[i1], 1); list[(size_t)i1 * Nc + s1] = t;
}

// ---------------------------------------------------------------------------
// MoE FFN (selected slots only). Token-row gather into LDS done by the
// Tensor Data Mover in gather mode (16-bit row indices) when available,
// otherwise by async global->LDS loads.
// grid = (Ec, Nc/16), block = 256 (8 waves)
// ---------------------------------------------------------------------------
__global__ void k_moe_ffn(const bf16_t* __restrict__ x2b, const float* __restrict__ w1,
                          const float* __restrict__ w2, const int* __restrict__ counts,
                          const int* __restrict__ list, float* __restrict__ feo) {
  const int e  = blockIdx.x;
  const int it = blockIdx.y;
  const int cnt = counts[e];
  if (it * 16 >= cnt) return;
  const int nrow = min(16, cnt - it * 16);

  __shared__ bf16_t xt[16 * Cc];            // 16 KB gathered token tile
  __shared__ bf16_t hid[16 * (Ic / 2)];     // 32 KB hidden slab
  __shared__ int tok[16];

  if (threadIdx.x < 16) {
    const int r = threadIdx.x;
    tok[r] = list[(size_t)e * Nc + it * 16 + min(r, nrow - 1)];
  }
  __syncthreads();

#if __has_builtin(__builtin_amdgcn_tensor_load_to_lds)
  if (threadIdx.x < 32) {   // wave 0 issues one TDM gather for the whole tile
    const unsigned lds_off = (unsigned)(uintptr_t)(void*)xt;
    const unsigned long long ga = (unsigned long long)(uintptr_t)(const void*)x2b;
    u32x4_t g0; i32x8_t g1; i32x4_t g2, g3;
    g0[0] = 1u | (1u << 31);                    // count=1, gather_mode, 16-bit idx
    g0[1] = lds_off;                            // D#.lds_addr
    g0[2] = (unsigned)(ga & 0xffffffffu);       // D#.global_addr lo
    g0[3] = (unsigned)((ga >> 32) & 0x01ffffffu) | (2u << 30);  // hi | type=2
    g1[0] = 1 << 16;                            // data_size = 2 bytes
    g1[1] = (Cc & 0xffff) << 16;                // tensor_dim0 = 512
    g1[2] = (Nc & 0xffff) << 16;                // tensor_dim1 = 2048
    g1[3] = (Cc & 0xffff) << 16;                // tile_dim0 = 512
    g1[4] = 16;                                 // tile_dim1 = #indices
    g1[5] = Cc;                                 // tensor_dim0_stride = 512
    g1[6] = 0; g1[7] = 0;
    g2[0] = (tok[0] & 0xffff) | (tok[1] << 16);
    g2[1] = (tok[2] & 0xffff) | (tok[3] << 16);
    g2[2] = (tok[4] & 0xffff) | (tok[5] << 16);
    g2[3] = (tok[6] & 0xffff) | (tok[7] << 16);
    g3[0] = (tok[8] & 0xffff) | (tok[9] << 16);
    g3[1] = (tok[10] & 0xffff) | (tok[11] << 16);
    g3[2] = (tok[12] & 0xffff) | (tok[13] << 16);
    g3[3] = (tok[14] & 0xffff) | (tok[15] << 16);
    i32x8_t g4 = {0, 0, 0, 0, 0, 0, 0, 0};      // unused extra descriptor group
    __builtin_amdgcn_tensor_load_to_lds(g0, g1, g2, g3, g4, 0);
    __builtin_amdgcn_s_wait_tensorcnt(0);
  }
  __syncthreads();
#else
  for (int idx = threadIdx.x; idx < (16 * Cc) / 8; idx += blockDim.x) {
    const int r = idx >> 6;          // Cc/8 = 64 16-byte chunks per row
    const int c8 = idx & 63;
    async_copy_b128(xt + idx * 8, x2b + (size_t)tok[r] * Cc + c8 * 8);
  }
  async_wait0();
  __syncthreads();
#endif

  const int wave = threadIdx.x >> 5;
  const int lane = threadIdx.x & 31;
  const int col  = lane & 15;
  const int r0   = (lane < 16) ? 0 : 8;
  v8f outacc[4];
#pragma unroll
  for (int j = 0; j < 4; ++j) outacc[j] = (v8f){};

  for (int half = 0; half < 2; ++half) {
    // phase 1: hidden[16, 1024] = gelu(xt @ w1 slab)
    for (int nt = wave; nt < (Ic / 2) / 16; nt += 8) {
      const int n0 = nt * 16;
      v8f acc = {};
      const float* W1 = w1 + (size_t)e * Cc * Ic + half * (Ic / 2) + n0;
      for (int k0 = 0; k0 < Cc; k0 += 32) {
        v16bf a = a_frag_bf16(xt, Cc, k0);
        v16bf b = b_frag_f32(W1, Ic, k0, 0);
        acc = wmma_bf16(a, b, acc);
      }
#pragma unroll
      for (int i = 0; i < 8; ++i) {
        const float x = acc[i];
        const float gl = 0.5f * x * (1.0f + erff(x * 0.70710678f));  // exact gelu
        hid[(r0 + i) * (Ic / 2) + n0 + col] = (bf16_t)gl;
      }
    }
    __syncthreads();
    // phase 2: out[16, C] += hidden @ w2 slab
#pragma unroll
    for (int j = 0; j < 4; ++j) {
      const int n0c = (wave * 4 + j) * 16;
      v8f acc = outacc[j];
      const float* W2 = w2 + (size_t)e * Ic * Cc + (size_t)(half * (Ic / 2)) * Cc + n0c;
      for (int k0 = 0; k0 < Ic / 2; k0 += 32) {
        v16bf a = a_frag_bf16(hid, Ic / 2, k0);
        v16bf b = b_frag_f32(W2, Cc, k0, 0);
        acc = wmma_bf16(a, b, acc);
      }
      outacc[j] = acc;
    }
    __syncthreads();
  }

#pragma unroll
  for (int j = 0; j < 4; ++j) {
    const int n0c = (wave * 4 + j) * 16;
#pragma unroll
    for (int i = 0; i < 8; ++i) {
      const int r = r0 + i;
      if (r < nrow)
        feo[((size_t)tok[r] * Ec + e) * Cc + n0c + col] = outacc[j][i];
    }
  }
}

// h += sum_k rw[t,k] * feo[t, sel[t,k], :] ; also emit bf16 copy for lm_head
__global__ void k_moe_combine(const float* __restrict__ feo, const float* __restrict__ rw,
                              const int* __restrict__ sel, float* __restrict__ h,
                              bf16_t* __restrict__ hb) {
  const int t = blockIdx.x;
  const float w0 = rw[t * 2], w1 = rw[t * 2 + 1];
  const int s0 = sel[t * 2], s1 = sel[t * 2 + 1];
  const float* f0 = feo + ((size_t)t * Ec + s0) * Cc;
  const float* f1 = feo + ((size_t)t * Ec + s1) * Cc;
  for (int c = threadIdx.x; c < Cc; c += blockDim.x) {
    const float nv = h[(size_t)t * Cc + c] + w0 * f0[c] + w1 * f1[c];
    h[(size_t)t * Cc + c] = nv;
    hb[(size_t)t * Cc + c] = (bf16_t)nv;
  }
}

// ---------------------------------------------------------------------------
// logits = h @ lm_head   [2048 x 512] @ [512 x 32000]; A staged in LDS async
// grid = (Vc/16/8, Nc/16), block = 256
// ---------------------------------------------------------------------------
__global__ void k_logits(const bf16_t* __restrict__ hb, const float* __restrict__ lm,
                         float* __restrict__ out) {
  __shared__ bf16_t at[16 * Cc];
  const int m0 = blockIdx.y * 16;
  stage_a_async(at, hb + (size_t)m0 * Cc);
  __syncthreads();

  const int wave = threadIdx.x >> 5;
  const int n0 = (blockIdx.x * 8 + wave) * 16;
  v8f acc = {};
#pragma unroll 4
  for (int k0 = 0; k0 < Cc; k0 += 32) {
    if (k0 + 32 < Cc)
      __builtin_prefetch(lm + (size_t)(k0 + 32) * Vc + n0, 0, 1);  // global_prefetch_b8
    v16bf a = a_frag_bf16(at, Cc, k0);
    v16bf b = b_frag_f32(lm, Vc, k0, n0);
    acc = wmma_bf16(a, b, acc);
  }
  store_tile_f32(out + (size_t)m0 * Vc + n0, Vc, acc);
}

// ---------------------------------------------------------------------------
extern "C" void kernel_launch(void* const* d_in, const int* in_sizes, int n_in,
                              void* d_out, int out_size, void* d_ws, size_t ws_size,
                              hipStream_t stream) {
  (void)in_sizes; (void)n_in; (void)out_size; (void)ws_size;
  const int*   ids  = (const int*)  d_in[0];
  const float* emb  = (const float*)d_in[1];
  const float* wq   = (const float*)d_in[2];
  const float* wk   = (const float*)d_in[3];
  const float* wv   = (const float*)d_in[4];
  const float* wo   = (const float*)d_in[5];
  const float* gate = (const float*)d_in[6];
  const float* w1   = (const float*)d_in[7];
  const float* w2   = (const float*)d_in[8];
  const float* g1   = (const float*)d_in[9];
  const float* b1   = (const float*)d_in[10];
  const float* g2   = (const float*)d_in[11];
  const float* b2   = (const float*)d_in[12];
  const float* lm   = (const float*)d_in[13];

  float* logits = (float*)d_out;                         // [N, V]
  float* feo    = logits + (size_t)Nc * Vc;              // [N, E, C]
  float* rl     = feo + (size_t)Nc * Ec * Cc;            // [N, E]
  float* x2     = rl + (size_t)Nc * Ec;                  // [N, C]

  char* wsp = (char*)d_ws;
  auto alloc = [&](size_t bytes) -> char* {
    char* p = wsp; wsp += (bytes + 255) & ~(size_t)255; return p;
  };
  float*  h    = (float*) alloc((size_t)Nc * Cc * 4);
  bf16_t* x1b  = (bf16_t*)alloc((size_t)Nc * Cc * 2);
  bf16_t* qb   = (bf16_t*)alloc((size_t)Nc * Cc * 2);
  bf16_t* kb   = (bf16_t*)alloc((size_t)Nc * Cc * 2);
  bf16_t* vb   = (bf16_t*)alloc((size_t)Nc * Cc * 2);
  bf16_t* aob  = (bf16_t*)alloc((size_t)Nc * Cc * 2);
  bf16_t* x2b  = (bf16_t*)alloc((size_t)Nc * Cc * 2);
  bf16_t* hb   = (bf16_t*)alloc((size_t)Nc * Cc * 2);
  float*  rw   = (float*) alloc((size_t)Nc * 2 * 4);
  int*    sel  = (int*)   alloc((size_t)Nc * 2 * 4);
  int*    cnts = (int*)   alloc((size_t)Ec * 4);
  int*    list = (int*)   alloc((size_t)Ec * Nc * 4);

  // zero full_expert_outputs (unselected slots must be zero) + expert counters
  k_zero_f32<<<4096, 256, 0, stream>>>(feo, (size_t)Nc * Ec * Cc);
  k_zero_i32<<<1, 64, 0, stream>>>(cnts, Ec);

  k_embed_ln1<<<Nc, 256, 0, stream>>>(ids, emb, g1, b1, h, x1b);
  k_qkv<<<dim3(Cc / 16 / 8, Nc / 16, 3), 256, 0, stream>>>(x1b, wq, wk, wv, qb, kb, vb);
  k_attn<<<dim3(Tc / 16, Bc * Hc), 32, 0, stream>>>(qb, kb, vb, aob);
  k_oproj<<<dim3(Cc / 16 / 8, Nc / 16), 256, 0, stream>>>(aob, wo, h);
  k_ln2<<<Nc, 256, 0, stream>>>(h, g2, b2, x2, x2b);
  k_router<<<Nc / 16, 64, 0, stream>>>(x2b, gate, rl);
  k_topk<<<Nc / 256, 256, 0, stream>>>(rl, rw, sel, cnts, list);
  k_moe_ffn<<<dim3(Ec, Nc / 16), 256, 0, stream>>>(x2b, w1, w2, cnts, list, feo);
  k_moe_combine<<<Nc, 256, 0, stream>>>(feo, rw, sel, h, hb);
  k_logits<<<dim3(Vc / 16 / 8, Nc / 16), 256, 0, stream>>>(hb, lm, logits);
}
